// WPGM_12730283065918
// MI455X (gfx1250) — compile-verified, matrix operand order checked by reading, not run on hardware
//
#include <hip/hip_runtime.h>
#include <hip/hip_bf16.h>

#define CDIM 256
#define HH   128
#define WW   128
#define NEMB 20
#define HW   (HH * WW)
#define BSTRIDE 20          // padded row stride (uints) of LDS B tile
#define BPX  (WW + 2)       // 130 pixels incl. halo
#define NSTG (16 * BPX)     // 2080 uints staged per (kc,ty)

typedef __attribute__((ext_vector_type(16))) __bf16 v16bf;
typedef __attribute__((ext_vector_type(8)))  float  v8f;

union Frag {
    uint4 q[2];
    v16bf v;
};

__device__ __forceinline__ unsigned int f32_to_bf16_bits(float f) {
    unsigned int u = __float_as_uint(f);
    return (u + 0x7FFFu + ((u >> 16) & 1u)) >> 16;   // RNE
}
__device__ __forceinline__ unsigned int pack_bf16(float lo, float hi) {
    return (f32_to_bf16_bits(lo) & 0xFFFFu) | (f32_to_bf16_bits(hi) << 16);
}

// ---------------------------------------------------------------------------
// Global average pool: one block per (b,c)
// ---------------------------------------------------------------------------
__global__ __launch_bounds__(256) void pool_kernel(const float* __restrict__ x,
                                                   float* __restrict__ pooled) {
    __shared__ float s[256];
    int bc = blockIdx.x;
    const float* p = x + (size_t)bc * HW;
    float sum = 0.f;
    for (int i = threadIdx.x; i < HW; i += 256) sum += p[i];
    s[threadIdx.x] = sum;
    __syncthreads();
    for (int off = 128; off > 0; off >>= 1) {
        if (threadIdx.x < off) s[threadIdx.x] += s[threadIdx.x + off];
        __syncthreads();
    }
    if (threadIdx.x == 0) pooled[bc] = s[0] * (1.0f / HW);
}

// ---------------------------------------------------------------------------
// Head: mapping net + gumbel quantize -> piror[b, 0..255]
// ---------------------------------------------------------------------------
__global__ __launch_bounds__(256) void head_kernel(const float* __restrict__ pooled,
                                                   const float* __restrict__ gumbel_u,
                                                   const float* __restrict__ W_map,
                                                   const float* __restrict__ proj_w,
                                                   const float* __restrict__ proj_b,
                                                   const float* __restrict__ embed,
                                                   float* __restrict__ piror) {
    __shared__ float ps[CDIM];
    __shared__ float ms[CDIM];
    __shared__ float lg[NEMB];
    __shared__ float fs[NEMB];
    int b = blockIdx.x;
    int c = threadIdx.x;

    ps[c] = pooled[b * CDIM + c];
    __syncthreads();

    float a = 0.f;
    for (int k = 0; k < CDIM; ++k) a += W_map[c * CDIM + k] * ps[k];
    ms[c] = 1.0f / (1.0f + __expf(-a));
    __syncthreads();

    if (c < NEMB) {
        float l = proj_b[c];
        for (int k = 0; k < CDIM; ++k) l += proj_w[c * CDIM + k] * ms[k];
        float u = gumbel_u[b * NEMB + c];
        float g = -__logf(-__logf(u + 1e-10f) + 1e-10f);
        lg[c] = l + g;   // TAU == 1
    }
    __syncthreads();

    if (c == 0) {
        float mx = lg[0]; int idx = 0;
        for (int n = 1; n < NEMB; ++n) if (lg[n] > mx) { mx = lg[n]; idx = n; }
        float e[NEMB]; float den = 0.f;
        for (int n = 0; n < NEMB; ++n) { e[n] = __expf(lg[n] - mx); den += e[n]; }
        for (int n = 0; n < NEMB; ++n) {
            float ys = e[n] / den;
            float yh = (n == idx) ? 1.0f : 0.0f;
            fs[n] = (yh + ys) - ys;   // straight-through forward value
        }
    }
    __syncthreads();

    float p = 0.f;
    for (int n = 0; n < NEMB; ++n) p += fs[n] * embed[n * CDIM + c];
    piror[b * CDIM + c] = p;
}

// ---------------------------------------------------------------------------
// Pack one conv layer's weights [256][256][3][3] fp32 into A-fragment order:
//   wp[ ((tap*8 + kc)*256 + m)*2 + kh ][0..7] = bf16 pairs of
//       A[m][kc*32 + kh*16 + 2p .. +2p+1]
// ---------------------------------------------------------------------------
__global__ __launch_bounds__(256) void pack_weights_kernel(const float* __restrict__ w,
                                                           unsigned int* __restrict__ wp) {
    int idx = blockIdx.x * 256 + threadIdx.x;   // 0 .. 294911
    int p   = idx & 7;
    int r   = idx >> 3;
    int kh  = r & 1;
    int rm  = r >> 1;
    int m   = rm & 255;
    int tk  = rm >> 8;         // t*8 + kc
    int kc  = tk & 7;
    int t   = tk >> 3;
    int k0  = kc * 32 + kh * 16 + 2 * p;
    const float* src = w + (size_t)m * (CDIM * 9) + (size_t)k0 * 9 + t;
    wp[idx] = pack_bf16(src[0], src[9]);
}

// ---------------------------------------------------------------------------
// h = x + piror : write fp32 h AND channel-pair-packed bf16 hpack
// ---------------------------------------------------------------------------
__global__ __launch_bounds__(256) void add_piror_pack_kernel(const float* __restrict__ x,
                                                             const float* __restrict__ piror,
                                                             float* __restrict__ h,
                                                             unsigned int* __restrict__ hpack) {
    size_t i = (size_t)blockIdx.x * 256 + threadIdx.x;
    int pix = (int)(i & (HW - 1));
    int bcp = (int)(i >> 14);
    int cp  = bcp & (CDIM / 2 - 1);
    int b   = bcp >> 7;
    size_t base = (size_t)(b * CDIM + 2 * cp) * HW + pix;
    float v0 = x[base]      + piror[b * CDIM + 2 * cp];
    float v1 = x[base + HW] + piror[b * CDIM + 2 * cp + 1];
    h[base]      = v0;
    h[base + HW] = v1;
    hpack[i] = pack_bf16(v0, v1);
}

// ---------------------------------------------------------------------------
// Staging helpers: global -> regs, regs -> LDS (double-buffered B tile)
// ---------------------------------------------------------------------------
__device__ __forceinline__ void stage_load(const unsigned int* __restrict__ inb,
                                           int y, int kc, int ty, int tid,
                                           unsigned int r[9]) {
    int yy = y + ty - 1;
    bool yok = (yy >= 0) && (yy < HH);
    #pragma unroll
    for (int j = 0; j < 9; ++j) {
        int idx = tid + j * 256;
        unsigned int v = 0u;
        if (idx < NSTG) {
            int kp = idx / BPX;
            int px = idx - kp * BPX;
            int xx = px - 1;
            if (yok && xx >= 0 && xx < WW)
                v = inb[(size_t)(kc * 16 + kp) * HW + yy * WW + xx];
        }
        r[j] = v;
    }
}

__device__ __forceinline__ void stage_store(unsigned int* buf, int tid,
                                            const unsigned int r[9]) {
    #pragma unroll
    for (int j = 0; j < 9; ++j) {
        int idx = tid + j * 256;
        if (idx < NSTG) {
            int kp = idx / BPX;
            int px = idx - kp * BPX;
            buf[px * BSTRIDE + kp] = r[j];
        }
    }
}

// ---------------------------------------------------------------------------
// Conv3x3 (pad=1) implicit GEMM, v_wmma_f32_16x16x32_bf16.
// Block = 256 threads = 8 waves (wm 0..3 x wn 0..1).
// Tile: 64 cout x 128 px (one full image row); 4 accumulators per wave.
// Double-buffered LDS B tile: one barrier per K-stage; next stage's global
// loads issue before the current stage's 12 WMMAs and land in LDS after.
// ---------------------------------------------------------------------------
__global__ __launch_bounds__(256) void conv3x3_wmma(const unsigned int* __restrict__ inpack,
                                                    const unsigned int* __restrict__ wpack,
                                                    const float* __restrict__ bias,
                                                    const float* __restrict__ skip,
                                                    float* __restrict__ out,
                                                    unsigned int* __restrict__ outpack,
                                                    int do_relu) {
    __shared__ unsigned int Blds[2][BPX * BSTRIDE];   // 2 x 10.4 KB

    int tid  = threadIdx.x;
    int lane = tid & 31;
    int wave = tid >> 5;
    int wm   = wave >> 1;     // 0..3 -> cout sub-tile
    int wn   = wave & 1;      // 0..1 -> 64-px half of the row

    int ptile    = blockIdx.x;         // 0..1023 : B * H
    int coutBase = blockIdx.y * 64;
    int b = ptile >> 7;
    int y = ptile & 127;

    const unsigned int* inb = inpack + (size_t)b * (CDIM / 2) * HW;

    v8f acc0 = {}, acc1 = {}, acc2 = {}, acc3 = {};
    int nn = lane & 15;
    int kh = lane >> 4;

    unsigned int rstg[9];

    // prologue: stage (kc=0, ty=0) into buffer 0
    stage_load(inb, y, 0, 0, tid, rstg);
    stage_store(Blds[0], tid, rstg);
    int p = 0;

    for (int kc = 0; kc < 8; ++kc) {
        for (int ty = 0; ty < 3; ++ty) {
            bool last = (kc == 7) && (ty == 2);
            int nkc = (ty == 2) ? kc + 1 : kc;
            int nty = (ty == 2) ? 0 : ty + 1;

            // issue next stage's global loads early (hide under WMMA burst)
            if (!last) stage_load(inb, y, nkc, nty, tid, rstg);

            __syncthreads();   // buf[p] staged; prior readers of buf[p^1] done

            // A fragments for this (kc,ty): 3 taps, register-resident
            Frag fa[3];
            #pragma unroll
            for (int tx = 0; tx < 3; ++tx) {
                int tap = ty * 3 + tx;
                const uint4* ap = (const uint4*)(wpack +
                    ((size_t)((tap * 8 + kc) * 256 + (coutBase + wm * 16 + nn)) * 2 + kh) * 8);
                fa[tx].q[0] = ap[0];
                fa[tx].q[1] = ap[1];
            }

            const unsigned int* buf = Blds[p];
            #pragma unroll
            for (int tx = 0; tx < 3; ++tx) {
                Frag fb0, fb1, fb2, fb3;
                int px0 = wn * 64 + nn + tx;   // staged index: x - (-1)
                const uint4* b0 = (const uint4*)(buf + (px0     ) * BSTRIDE + kh * 8);
                const uint4* b1 = (const uint4*)(buf + (px0 + 16) * BSTRIDE + kh * 8);
                const uint4* b2 = (const uint4*)(buf + (px0 + 32) * BSTRIDE + kh * 8);
                const uint4* b3 = (const uint4*)(buf + (px0 + 48) * BSTRIDE + kh * 8);
                fb0.q[0] = b0[0]; fb0.q[1] = b0[1];
                fb1.q[0] = b1[0]; fb1.q[1] = b1[1];
                fb2.q[0] = b2[0]; fb2.q[1] = b2[1];
                fb3.q[0] = b3[0]; fb3.q[1] = b3[1];
                acc0 = __builtin_amdgcn_wmma_f32_16x16x32_bf16(
                           false, fa[tx].v, false, fb0.v, (short)0, acc0, false, false);
                acc1 = __builtin_amdgcn_wmma_f32_16x16x32_bf16(
                           false, fa[tx].v, false, fb1.v, (short)0, acc1, false, false);
                acc2 = __builtin_amdgcn_wmma_f32_16x16x32_bf16(
                           false, fa[tx].v, false, fb2.v, (short)0, acc2, false, false);
                acc3 = __builtin_amdgcn_wmma_f32_16x16x32_bf16(
                           false, fa[tx].v, false, fb3.v, (short)0, acc3, false, false);
            }

            // write next stage's data into the other buffer (after frag loads)
            if (!last) stage_store(Blds[p ^ 1], tid, rstg);
            p ^= 1;
        }
    }

    // ---- epilogue ----
    float bvals[8];
    #pragma unroll
    for (int r = 0; r < 8; ++r) bvals[r] = bias[coutBase + wm * 16 + 8 * kh + r];

    #pragma unroll
    for (int sub = 0; sub < 4; ++sub) {
        v8f a = (sub == 0) ? acc0 : (sub == 1) ? acc1 : (sub == 2) ? acc2 : acc3;
        int x = wn * 64 + sub * 16 + nn;
        float v[8];
        #pragma unroll
        for (int r = 0; r < 8; ++r) {
            int cout = coutBase + wm * 16 + 8 * kh + r;
            float t = a[r] + bvals[r];
            if (do_relu) t = fmaxf(t, 0.0f);
            size_t o = ((size_t)(b * CDIM + cout) * HH + y) * WW + x;
            if (skip) t += skip[o];
            if (out) out[o] = t;
            v[r] = t;
        }
        #pragma unroll
        for (int j = 0; j < 4; ++j) {
            int cpg = coutBase / 2 + wm * 8 + kh * 4 + j;
            size_t op = ((size_t)(b * (CDIM / 2) + cpg) * HH + y) * WW + x;
            outpack[op] = pack_bf16(v[2 * j], v[2 * j + 1]);
        }
    }
}

// ---------------------------------------------------------------------------
extern "C" void kernel_launch(void* const* d_in, const int* in_sizes, int n_in,
                              void* d_out, int out_size, void* d_ws, size_t ws_size,
                              hipStream_t stream) {
    const float* x        = (const float*)d_in[0];
    const float* gumbel_u = (const float*)d_in[1];
    const float* W_map    = (const float*)d_in[2];
    const float* proj_w   = (const float*)d_in[3];
    const float* proj_b   = (const float*)d_in[4];
    const float* embed    = (const float*)d_in[5];
    const float* res_w1   = (const float*)d_in[6];
    const float* res_b1   = (const float*)d_in[7];
    const float* res_w2   = (const float*)d_in[8];
    const float* res_b2   = (const float*)d_in[9];

    float* h = (float*)d_out;   // fp32 h lives in d_out

    const size_t pack_elems  = (size_t)8 * (CDIM / 2) * HW;        // 16.7M uints
    const size_t wpack_elems = (size_t)9 * 8 * 256 * 2 * 8;        // 294912 uints/layer
    unsigned int* hpack  = (unsigned int*)d_ws;
    unsigned int* tpack  = hpack + pack_elems;
    unsigned int* wpacks = tpack + pack_elems;                     // 6 layers
    float* pooled = (float*)(wpacks + 6 * wpack_elems);
    float* piror  = pooled + 8 * CDIM;

    for (int l = 0; l < 6; ++l) {
        int i = l >> 1;
        const float* wsrc = (l & 1) ? (res_w2 + (size_t)i * CDIM * CDIM * 9)
                                    : (res_w1 + (size_t)i * CDIM * CDIM * 9);
        pack_weights_kernel<<<(unsigned)(wpack_elems / 256), 256, 0, stream>>>(
            wsrc, wpacks + (size_t)l * wpack_elems);
    }

    pool_kernel<<<8 * CDIM, 256, 0, stream>>>(x, pooled);
    head_kernel<<<8, 256, 0, stream>>>(pooled, gumbel_u, W_map, proj_w, proj_b,
                                       embed, piror);
    add_piror_pack_kernel<<<(unsigned)(pack_elems / 256), 256, 0, stream>>>(
        x, piror, h, hpack);

    dim3 grid(8 * HH, CDIM / 64);   // (1024, 4): one full row per block
    for (int i = 0; i < 3; ++i) {
        const unsigned int* w1p = wpacks + (size_t)(2 * i) * wpack_elems;
        const unsigned int* w2p = wpacks + (size_t)(2 * i + 1) * wpack_elems;
        const float* b1 = res_b1 + i * CDIM;
        const float* b2 = res_b2 + i * CDIM;
        conv3x3_wmma<<<grid, 256, 0, stream>>>(hpack, w1p, b1, nullptr, nullptr,
                                               tpack, 1);
        conv3x3_wmma<<<grid, 256, 0, stream>>>(tpack, w2p, b2, h, h, hpack, 0);
    }
}
